// Model_87101936763051
// MI455X (gfx1250) — matrix operand order repaired and checked by reading, not compiled, not run on hardware
//
#include <hip/hip_runtime.h>

// MI455X / gfx1250, wave32. All matmuls via v_wmma_f32_16x16x32_bf16 with
// fragment loads as ds_load_b128 pairs; fp32->bf16 via v_cvt_pk_bf16_f32
// where the toolchain declares it (manual RNE fallback otherwise).

typedef __attribute__((ext_vector_type(16))) __bf16 bf16x16;
typedef __attribute__((ext_vector_type(8)))  __bf16 bf16x8;
typedef __attribute__((ext_vector_type(2)))  __bf16 bf16x2;
typedef __attribute__((ext_vector_type(8)))  float  f32x8;
typedef __attribute__((ext_vector_type(4)))  float  f32x4;
typedef __attribute__((ext_vector_type(2)))  float  f32x2;
typedef __attribute__((ext_vector_type(8)))  unsigned u32x8;

#define BATCH 128
#define NSEQ  256
#define HEADS 7
#define CDIM  128

__device__ __forceinline__ unsigned bfbits(float f) {   // fp32 -> bf16 bits, RNE
  unsigned u = __builtin_bit_cast(unsigned, f);
  return (u + 0x7FFFu + ((u >> 16) & 1u)) >> 16;
}
__device__ __forceinline__ unsigned pk2(float lo, float hi) {
#if __has_builtin(__builtin_amdgcn_cvt_pk_bf16_f32)
  bf16x2 r = __builtin_amdgcn_cvt_pk_bf16_f32(lo, hi);   // D.lo=cvt(S0), D.hi=cvt(S1)
  return __builtin_bit_cast(unsigned, r);
#else
  return bfbits(lo) | (bfbits(hi) << 16);
#endif
}
__device__ __forceinline__ f32x8 zero8() {
  f32x8 z = {0.f, 0.f, 0.f, 0.f, 0.f, 0.f, 0.f, 0.f};
  return z;
}
__device__ __forceinline__ bf16x16 cat8(bf16x8 a, bf16x8 b) {
  return __builtin_shufflevector(a, b, 0, 1, 2, 3, 4, 5, 6, 7,
                                 8, 9, 10, 11, 12, 13, 14, 15);
}
__device__ __forceinline__ f32x8 wmma_bf16(bf16x16 a, bf16x16 b, f32x8 c) {
  return __builtin_amdgcn_wmma_f32_16x16x32_bf16(false, a, false, b, (short)0, c,
                                                 false, false);
}

// ---------------------------------------------------------------------------
// Fused GEMM: Y[z] = epi( W[BM x K] @ X[z][K x Ntile] ); bias/BN/ReLU fused.
// LDS rows are 40 bf16 (80 B) so every fragment run is 16-byte aligned.
// grid = (N/BN, 1, batch); block = (BM/16)*32.
// ---------------------------------------------------------------------------
template <int BM, int BN>
__global__ void k_gemm_bn(const float* __restrict__ W, int K,
                          const float* __restrict__ X, long strideX, int ldX,
                          float* __restrict__ Y, long strideY, int ldY,
                          const float* __restrict__ bias, int bias_div,
                          const float* __restrict__ gamma,
                          const float* __restrict__ beta,
                          const float* __restrict__ mean,
                          const float* __restrict__ var,
                          int do_relu) {
  constexpr int NT = (BM / 16) * 32;
  __shared__ __attribute__((aligned(16))) unsigned As32[BM][20];  // [m][k] k-contig
  __shared__ __attribute__((aligned(16))) unsigned Bs32[BN][20];  // [n][k] k-contig

  const int tid  = threadIdx.x;
  const int wv   = tid >> 5;
  const int lane = tid & 31;
  const int half = lane >> 4;
  const int lr   = lane & 15;
  const int n0   = blockIdx.x * BN;
  const long xb  = (long)blockIdx.z * strideX;
  const long yb  = (long)blockIdx.z * strideY;

  f32x8 acc[BN / 16];
#pragma unroll
  for (int t = 0; t < BN / 16; ++t) acc[t] = zero8();

  for (int k0 = 0; k0 < K; k0 += 32) {
    // stage A: pack bf16 pairs along k (float2 global loads, b32 LDS stores)
    for (int i = tid; i < BM * 16; i += NT) {
      int r = i >> 4, c = i & 15;
      const f32x2 wp = *(const f32x2*)(W + (long)r * K + k0 + 2 * c);
      As32[r][c] = pk2(wp[0], wp[1]);
    }
    // stage B transposed: Bs32[n][k], coalesced over n
    for (int i = tid; i < BN * 16; i += NT) {
      int nl = i & (BN - 1), c = i >> 6;
      const float* xp = X + xb + (long)(k0 + 2 * c) * ldX + n0 + nl;
      Bs32[nl][c] = pk2(xp[0], xp[ldX]);
    }
    if (k0 + 32 < K)
      __builtin_prefetch(&X[xb + (long)(k0 + 32) * ldX + n0 + (tid & 63)], 0, 1);
    __syncthreads();

    // A fragment: two contiguous 16B runs (K = 8h..8h+7, 8h+16..8h+23)
    const bf16x8* pa = (const bf16x8*)As32[(wv << 4) + lr];
    bf16x16 af = cat8(pa[half], pa[half + 2]);

#pragma unroll
    for (int t = 0; t < BN / 16; ++t) {
      const bf16x8* pb = (const bf16x8*)Bs32[(t << 4) + lr];
      bf16x16 bfr = cat8(pb[half], pb[half + 2]);
      acc[t] = wmma_bf16(af, bfr, acc[t]);
    }
    __syncthreads();
  }

#pragma unroll
  for (int t = 0; t < BN / 16; ++t) {
#pragma unroll
    for (int r = 0; r < 8; ++r) {
      const int m = (wv << 4) + r + (half << 3);
      const int n = n0 + (t << 4) + lr;
      float val = acc[t][r];
      if (bias)  val += bias[bias_div ? (n / bias_div) : m];
      if (gamma) val = (val - mean[m]) * (gamma[m] * rsqrtf(var[m] + 1e-5f)) + beta[m];
      if (do_relu) val = fmaxf(val, 0.f);
      Y[yb + (long)m * ldY + n] = val;   // coalesced across lanes
    }
  }
}

// ---------------------------------------------------------------------------
// Fused window attention, one (b, h, 64-query block); 4 waves x 16 queries.
// GEMM1 computes S^T = kwin^T(A) @ qwin(B): lane owns one query column n=lr,
// softmax over m needs a single shfl_xor(16). The exp'd accumulators are
// *already* the A-fragment layout for GEMM2 (O^T = P @ v^T): lane holds row
// M = n = lr with K-residues [8*half, 8*half+8) mod 16 -- no LDS round-trip.
// ---------------------------------------------------------------------------
__global__ void __launch_bounds__(128)
k_attn(const float* __restrict__ q, const float* __restrict__ k,
       const float* __restrict__ v, float* __restrict__ O) {
  __shared__ __attribute__((aligned(16))) unsigned Qt[64][20];    // [n][w] w-contig
  __shared__ __attribute__((aligned(16))) unsigned Kt[256][20];   // [m][w] w-contig
  __shared__ __attribute__((aligned(16))) unsigned Vs[32][132];   // [w][m] m-contig

  const int tid  = threadIdx.x;
  const int wv   = tid >> 5;
  const int lane = tid & 31;
  const int half = lane >> 4;
  const int lr   = lane & 15;
  const int n0   = blockIdx.x * 64;
  const int b    = blockIdx.y / HEADS;
  const int h    = blockIdx.y % HEADS;
  const long base = ((long)b * CDIM + h * 16) * NSEQ;  // window ch0 = h*16

  for (int i = tid; i < 64 * 16; i += 128) {     // Qt[n][w]: pack along w
    int nl = i & 63, w2 = i >> 6;
    const float* qp = q + base + (long)(2 * w2) * NSEQ + n0 + nl;
    Qt[nl][w2] = pk2(qp[0], qp[NSEQ]);
  }
  for (int i = tid; i < 256 * 16; i += 128) {    // Kt[m][w]: pack along w
    int m = i & 255, w2 = i >> 8;
    const float* kp = k + base + (long)(2 * w2) * NSEQ + m;
    Kt[m][w2] = pk2(kp[0], kp[NSEQ]);
  }
  for (int i = tid; i < 32 * 128; i += 128) {    // Vs[w][m]: float2-coalesced
    int m2 = i & 127, w = i >> 7;
    const f32x2 vv = *(const f32x2*)(v + base + (long)w * NSEQ + 2 * m2);
    Vs[w][m2] = pk2(vv[0], vv[1]);
  }
  __syncthreads();

  // GEMM1: 16 m-stripes, K = 32 (one WMMA step each). B frag is loop-invariant.
  const bf16x8* pq = (const bf16x8*)Qt[(wv << 4) + lr];
  const bf16x16 bq = cat8(pq[half], pq[half + 2]);
  f32x8 acc[16];
#pragma unroll
  for (int s = 0; s < 16; ++s) {
    const bf16x8* pk_ = (const bf16x8*)Kt[(s << 4) + lr];
    bf16x16 ak = cat8(pk_[half], pk_[half + 2]);
    acc[s] = wmma_bf16(ak, bq, zero8());
  }

  // Softmax over m (lane-local 128 values + one cross-half exchange).
  float mx = acc[0][0];
#pragma unroll
  for (int s = 0; s < 16; ++s)
#pragma unroll
    for (int r = 0; r < 8; ++r) mx = fmaxf(mx, acc[s][r]);
  mx = fmaxf(mx, __shfl_xor(mx, 16, 32));
  float sum = 0.f;
#pragma unroll
  for (int s = 0; s < 16; ++s)
#pragma unroll
    for (int r = 0; r < 8; ++r) {
      float e = __expf(acc[s][r] - mx);
      acc[s][r] = e;
      sum += e;
    }
  sum += __shfl_xor(sum, 16, 32);
  const float inv = 1.f / sum;

  // Re-pack accumulators directly as the 8 A-fragments of GEMM2
  // (packed converts: 8 x v_cvt_pk_bf16_f32 per fragment).
  bf16x16 ap[8];
#pragma unroll
  for (int s2 = 0; s2 < 8; ++s2) {
    u32x8 t;
#pragma unroll
    for (int jj = 0; jj < 8; ++jj) {
      const int s = 2 * s2 + (jj >= 4);
      const int r0 = (2 * jj) & 7;
      t[jj] = pk2(acc[s][r0] * inv, acc[s][r0 + 1] * inv);
    }
    ap[s2] = __builtin_bit_cast(bf16x16, t);
  }

  // GEMM2: O^T = P @ v^T; B frags from Vs rows (m-contiguous).
  f32x8 o0 = zero8(), o1 = zero8();
  const bf16x8* pv0 = (const bf16x8*)Vs[lr];
  const bf16x8* pv1 = (const bf16x8*)Vs[16 + lr];
#pragma unroll
  for (int s2 = 0; s2 < 8; ++s2) {
    const int ch = (s2 << 2) | half;              // elem 32*s2 + 8*half (+16)
    o0 = wmma_bf16(ap[s2], cat8(pv0[ch], pv0[ch + 2]), o0);
    o1 = wmma_bf16(ap[s2], cat8(pv1[ch], pv1[ch + 2]), o1);
  }

  // D layout gives each lane 8 consecutive output floats -> b128 stores.
  const long ob = ((long)b * (HEADS * 32) + h * 32) * NSEQ;
  float* po0 = O + ob + (long)lr * NSEQ + n0 + (wv << 4) + (half << 3);
  float* po1 = po0 + (long)16 * NSEQ;
  *(f32x4*)po0       = __builtin_shufflevector(o0, o0, 0, 1, 2, 3);
  *((f32x4*)po0 + 1) = __builtin_shufflevector(o0, o0, 4, 5, 6, 7);
  *(f32x4*)po1       = __builtin_shufflevector(o1, o1, 0, 1, 2, 3);
  *((f32x4*)po1 + 1) = __builtin_shufflevector(o1, o1, 4, 5, 6, 7);
}

// frep broadcast (b128 stores): xcat[b][128+i][4t..4t+3] = feat[b][i]
__global__ void k_frep(const float* __restrict__ feat, float* __restrict__ xcat) {
  long idx = (long)blockIdx.x * blockDim.x + threadIdx.x;
  if (idx >= (long)BATCH * 512 * 64) return;
  int  t4 = (int)(idx & 63);
  long bi = idx >> 6;                 // b*512 + i
  int  i  = (int)(bi & 511);
  int  b  = (int)(bi >> 9);
  float f = feat[bi];
  f32x4 vv = {f, f, f, f};
  *(f32x4*)(xcat + ((long)b * 640 + 128 + i) * NSEQ + 4 * t4) = vv;
}

// out[b][c][n] = b2[c] + sum_j W2[c][j] * h[b][j][n], vectorized over n.
__global__ void k_head(const float* __restrict__ h, const float* __restrict__ W2,
                       const float* __restrict__ b2, float* __restrict__ out) {
  int idx = blockIdx.x * blockDim.x + threadIdx.x;
  if (idx >= BATCH * 3 * 64) return;
  int n4 = idx & 63;
  int c  = (idx >> 6) % 3;
  int b  = idx / (3 * 64);
  float bb = b2[c];
  f32x4 s = {bb, bb, bb, bb};
  const float* hb = h + (long)b * 64 * NSEQ + 4 * n4;
#pragma unroll 8
  for (int j = 0; j < 64; ++j)
    s += W2[c * 64 + j] * (*(const f32x4*)(hb + (long)j * NSEQ));
  *(f32x4*)(out + ((long)b * 3 + c) * NSEQ + 4 * n4) = s;
}

// ---------------------------------------------------------------------------
// Input flattening (setup_inputs dict insertion order, depth-first):
//  0 feat | 1 W_ps | 2 b_ps | 3.. tf1 (21) | 24.. tf2 (21) | 45.. tf3 (21)
//  66 W1 | 67 b1 | 68 W2 | 69 b2
// ---------------------------------------------------------------------------
extern "C" void kernel_launch(void* const* d_in, const int* in_sizes, int n_in,
                              void* d_out, int out_size, void* d_ws, size_t ws_size,
                              hipStream_t stream) {
  (void)in_sizes; (void)n_in; (void)out_size; (void)ws_size;

  const float* feat = (const float*)d_in[0];
  const float* Wps  = (const float*)d_in[1];
  const float* bps  = (const float*)d_in[2];
  const int TF1 = 3, TF2 = 24, TF3 = 45;
  const float* W1 = (const float*)d_in[66];
  const float* b1 = (const float*)d_in[67];
  const float* W2 = (const float*)d_in[68];
  const float* b2 = (const float*)d_in[69];

  float* xcat = (float*)d_ws;                         // B x 640 x 256
  float* qb = xcat + (size_t)BATCH * 640 * NSEQ;      // B x 128 x 256
  float* kb = qb + (size_t)BATCH * CDIM * NSEQ;
  float* vb = kb + (size_t)BATCH * CDIM * NSEQ;
  float* Ob = vb + (size_t)BATCH * CDIM * NSEQ;       // B x 224 x 256
  float* xb = Ob + (size_t)BATCH * 224 * NSEQ;        // B x 128 x 256 (x1/x3)
  float* hb = xb + (size_t)BATCH * CDIM * NSEQ;       // B x 64  x 256

  {
    long n = (long)BATCH * 512 * 64;
    k_frep<<<dim3((unsigned)((n + 255) / 256)), dim3(256), 0, stream>>>(feat, xcat);
  }
  // x0 = feat(128x512) @ W_ps(512x32768) + b_ps[n/256] -> xcat channels 0..127
  k_gemm_bn<128, 64><<<dim3(32768 / 64, 1, 1), dim3(256), 0, stream>>>(
      feat, 512, Wps, 0L, 32768, xcat, 0L, 640 * NSEQ,
      bps, NSEQ, nullptr, nullptr, nullptr, nullptr, 0);

  auto scmh = [&](const float* xin, long sxin, int cin, int pbase,
                  float* yout, long syout, int ldy) {
    const float* Wq  = (const float*)d_in[pbase + 0];
    const float* Wk  = (const float*)d_in[pbase + 5];
    const float* Wv  = (const float*)d_in[pbase + 10];
    const float* Wf  = (const float*)d_in[pbase + 15];
    const float* bf_ = (const float*)d_in[pbase + 16];
    const float* bn_[4][4];
    const int boff[4] = {pbase + 1, pbase + 6, pbase + 11, pbase + 17};
    for (int s = 0; s < 4; ++s)
      for (int j = 0; j < 4; ++j) bn_[s][j] = (const float*)d_in[boff[s] + j];

    dim3 g4(NSEQ / 64, 1, BATCH);
    k_gemm_bn<128, 64><<<g4, 256, 0, stream>>>(
        Wq, cin, xin, sxin, NSEQ, qb, (long)CDIM * NSEQ, NSEQ,
        nullptr, 0, bn_[0][0], bn_[0][1], bn_[0][2], bn_[0][3], 1);
    k_gemm_bn<128, 64><<<g4, 256, 0, stream>>>(
        Wk, cin, xin, sxin, NSEQ, kb, (long)CDIM * NSEQ, NSEQ,
        nullptr, 0, bn_[1][0], bn_[1][1], bn_[1][2], bn_[1][3], 1);
    k_gemm_bn<128, 64><<<g4, 256, 0, stream>>>(
        Wv, cin, xin, sxin, NSEQ, vb, (long)CDIM * NSEQ, NSEQ,
        nullptr, 0, bn_[2][0], bn_[2][1], bn_[2][2], bn_[2][3], 1);
    k_attn<<<dim3(NSEQ / 64, BATCH * HEADS), dim3(128), 0, stream>>>(qb, kb, vb, Ob);
    k_gemm_bn<128, 64><<<g4, 256, 0, stream>>>(
        Wf, 224, Ob, (long)224 * NSEQ, NSEQ, yout, syout, ldy,
        bf_, 0, bn_[3][0], bn_[3][1], bn_[3][2], bn_[3][3], 1);
  };

  scmh(xcat, (long)640 * NSEQ, 640, TF1, xb, (long)CDIM * NSEQ, NSEQ);   // x1
  scmh(xb, (long)CDIM * NSEQ, CDIM, TF2, xcat, (long)640 * NSEQ, NSEQ);  // x2
  scmh(xcat, (long)640 * NSEQ, 640, TF3, xb, (long)CDIM * NSEQ, NSEQ);   // x3

  k_gemm_bn<64, 64><<<dim3(NSEQ / 64, 1, BATCH), dim3(128), 0, stream>>>(
      W1, CDIM, xb, (long)CDIM * NSEQ, NSEQ, hb, (long)64 * NSEQ, NSEQ,
      b1, 0, nullptr, nullptr, nullptr, nullptr, 1);
  k_head<<<dim3((BATCH * 3 * 64 + 255) / 256), dim3(256), 0, stream>>>(
      hb, W2, b2, (float*)d_out);
}
